// EventAugmentedLSTMCell_21294447853992
// MI455X (gfx1250) — matrix operand order repaired and checked by reading, not compiled, not run on hardware
//
#include <hip/hip_runtime.h>
#include <hip/hip_bf16.h>

// Problem dims (fixed by the reference)
#define B_SZ 4096
#define D_SZ 512
#define H_SZ 512
#define S_SZ 32

typedef __attribute__((ext_vector_type(16))) __bf16 v16bf;
typedef __attribute__((ext_vector_type(8)))  __bf16 v8bf;
typedef __attribute__((ext_vector_type(8)))  float  v8f;
typedef __attribute__((ext_vector_type(4)))  float  v4f;

// ---------------------------------------------------------------------------
// Generic f32 -> bf16 conversion (vectorized 4-wide). n must be multiple of 4.
// ---------------------------------------------------------------------------
__global__ __launch_bounds__(256) void cvt_bf16_kernel(const float* __restrict__ in,
                                                       __bf16* __restrict__ out,
                                                       int n4) {
    int i = blockIdx.x * 256 + threadIdx.x;
    if (i >= n4) return;
    v4f v = *(const v4f*)(in + (size_t)i * 4);
    __bf16* o = out + (size_t)i * 4;
    o[0] = (__bf16)v.x; o[1] = (__bf16)v.y; o[2] = (__bf16)v.z; o[3] = (__bf16)v.w;
}

// ---------------------------------------------------------------------------
// Kernel 0: softmax over mem_weights (32) and P[d] = sum_s w[s]*pos[s,d]
// ---------------------------------------------------------------------------
__global__ void prep_kernel(const float* __restrict__ mw,
                            const float* __restrict__ pos,
                            float* __restrict__ w_out,
                            float* __restrict__ P_out) {
    int d = threadIdx.x;  // 0..511
    float mx = -1e30f;
    #pragma unroll
    for (int s = 0; s < S_SZ; ++s) mx = fmaxf(mx, mw[s]);
    float e[S_SZ];
    float sum = 0.f;
    #pragma unroll
    for (int s = 0; s < S_SZ; ++s) { e[s] = __expf(mw[s] - mx); sum += e[s]; }
    float inv = 1.f / sum;
    if (d < S_SZ) w_out[d] = e[d] * inv;
    float acc = 0.f;
    #pragma unroll
    for (int s = 0; s < S_SZ; ++s) acc += e[s] * inv * pos[s * D_SZ + d];
    P_out[d] = acc;
}

// ---------------------------------------------------------------------------
// Kernel 1: event detector. One wave32 per batch row.
// ---------------------------------------------------------------------------
__global__ __launch_bounds__(128) void detector_kernel(
        const float* __restrict__ x,
        const float* __restrict__ dw,
        const float* __restrict__ db,
        const int*   __restrict__ ptr,
        int* __restrict__ mask_out,
        int* __restrict__ ptr_new_out) {
    const int wave = threadIdx.x >> 5;
    const int lane = threadIdx.x & 31;
    const int b = blockIdx.x * 4 + wave;
    const float* xr = x + (size_t)b * D_SZ;
    float s = 0.f;
    #pragma unroll 4
    for (int k = lane; k < D_SZ; k += 32) s += xr[k] * dw[k];
    #pragma unroll
    for (int off = 16; off >= 1; off >>= 1) s += __shfl_xor(s, off, 32);
    if (lane == 0) {
        float z  = s + db[0];
        float sg = 1.f / (1.f + __expf(-z));
        int m = sg > 0.5f ? 1 : 0;          // TAU = 0.5
        mask_out[b]    = m;
        ptr_new_out[b] = (ptr[b] + m) % S_SZ;
    }
}

// ---------------------------------------------------------------------------
// WMMA accumulate helper: adds A[32 rows @ mTile] * W^T[64 cols @ nTile] into
// 8 accumulators (2 M-subtiles x 4 N-subtiles), K-major bf16 operands.
// Each k-step: 2 A fragments + 4 B fragments (12 x b128 loads) feed 8
// v_wmma_f32_16x16x32_bf16 ops; B fragments are reused across both
// M-subtiles to raise FLOP/byte out of L2.
// Fragment layouts used here: A (16x32, 16-bit): lane holds row,
// K = 16*half + e -> two adjacent 8-element chunks; B (32x16): lane holds col,
// K = 16*half + e -> 16 contiguous elements.  16B-aligned -> global_load_b128.
// ---------------------------------------------------------------------------
__device__ __forceinline__ void gemm_accum(
        const __bf16* __restrict__ A, int lda,
        const __bf16* __restrict__ W, int ldw,
        int K, int mTile, int nTile, int l15, int half,
        v8f (&acc)[8]) {
    const __bf16* __restrict__ arow0 = A + (size_t)(mTile + l15) * lda;
    const __bf16* __restrict__ arow1 = A + (size_t)(mTile + 16 + l15) * lda;
    const __bf16* __restrict__ wr0 = W + (size_t)(nTile + 0  + l15) * ldw + 16 * half;
    const __bf16* __restrict__ wr1 = W + (size_t)(nTile + 16 + l15) * ldw + 16 * half;
    const __bf16* __restrict__ wr2 = W + (size_t)(nTile + 32 + l15) * ldw + 16 * half;
    const __bf16* __restrict__ wr3 = W + (size_t)(nTile + 48 + l15) * ldw + 16 * half;

    #pragma unroll 2
    for (int k0 = 0; k0 < K; k0 += 32) {
        __builtin_prefetch(arow0 + k0 + 64, 0, 1);
        __builtin_prefetch(arow1 + k0 + 64, 0, 1);

        v8bf a0lo = *(const v8bf*)(arow0 + k0 + 16 * half);
        v8bf a0hi = *(const v8bf*)(arow0 + k0 + 8 + 16 * half);
        v8bf a1lo = *(const v8bf*)(arow1 + k0 + 16 * half);
        v8bf a1hi = *(const v8bf*)(arow1 + k0 + 8 + 16 * half);
        v16bf a0 = __builtin_shufflevector(a0lo, a0hi,
                      0, 1, 2, 3, 4, 5, 6, 7, 8, 9, 10, 11, 12, 13, 14, 15);
        v16bf a1 = __builtin_shufflevector(a1lo, a1hi,
                      0, 1, 2, 3, 4, 5, 6, 7, 8, 9, 10, 11, 12, 13, 14, 15);

        v8bf b0lo = *(const v8bf*)(wr0 + k0);
        v8bf b0hi = *(const v8bf*)(wr0 + k0 + 8);
        v8bf b1lo = *(const v8bf*)(wr1 + k0);
        v8bf b1hi = *(const v8bf*)(wr1 + k0 + 8);
        v8bf b2lo = *(const v8bf*)(wr2 + k0);
        v8bf b2hi = *(const v8bf*)(wr2 + k0 + 8);
        v8bf b3lo = *(const v8bf*)(wr3 + k0);
        v8bf b3hi = *(const v8bf*)(wr3 + k0 + 8);

        v16bf b0 = __builtin_shufflevector(b0lo, b0hi,
                      0, 1, 2, 3, 4, 5, 6, 7, 8, 9, 10, 11, 12, 13, 14, 15);
        v16bf b1 = __builtin_shufflevector(b1lo, b1hi,
                      0, 1, 2, 3, 4, 5, 6, 7, 8, 9, 10, 11, 12, 13, 14, 15);
        v16bf b2 = __builtin_shufflevector(b2lo, b2hi,
                      0, 1, 2, 3, 4, 5, 6, 7, 8, 9, 10, 11, 12, 13, 14, 15);
        v16bf b3 = __builtin_shufflevector(b3lo, b3hi,
                      0, 1, 2, 3, 4, 5, 6, 7, 8, 9, 10, 11, 12, 13, 14, 15);

        acc[0] = __builtin_amdgcn_wmma_f32_16x16x32_bf16(false, a0, false, b0,
                                                         (short)0, acc[0], false, false);
        acc[1] = __builtin_amdgcn_wmma_f32_16x16x32_bf16(false, a0, false, b1,
                                                         (short)0, acc[1], false, false);
        acc[2] = __builtin_amdgcn_wmma_f32_16x16x32_bf16(false, a0, false, b2,
                                                         (short)0, acc[2], false, false);
        acc[3] = __builtin_amdgcn_wmma_f32_16x16x32_bf16(false, a0, false, b3,
                                                         (short)0, acc[3], false, false);
        acc[4] = __builtin_amdgcn_wmma_f32_16x16x32_bf16(false, a1, false, b0,
                                                         (short)0, acc[4], false, false);
        acc[5] = __builtin_amdgcn_wmma_f32_16x16x32_bf16(false, a1, false, b1,
                                                         (short)0, acc[5], false, false);
        acc[6] = __builtin_amdgcn_wmma_f32_16x16x32_bf16(false, a1, false, b2,
                                                         (short)0, acc[6], false, false);
        acc[7] = __builtin_amdgcn_wmma_f32_16x16x32_bf16(false, a1, false, b3,
                                                         (short)0, acc[7], false, false);
    }
}

// ---------------------------------------------------------------------------
// Kernel 2: WMMA bf16 GEMM  C[M,N] = sum_p A_p * W_p^T (+ bias).
// 128 threads = 4 waves; each wave owns a 32(M) x 64(N) output tile.
// Pointers are passed as direct kernel args so address-space inference keeps
// every operand load on the GLOBAL path (no flat_load / DScnt coupling).
// Optional float and/or bf16 outputs (bf16 feeds downstream GEMMs).
// ---------------------------------------------------------------------------
__global__ __launch_bounds__(128) void wmma_gemm_kernel(
        const __bf16* __restrict__ A0, int lda0, const __bf16* __restrict__ W0, int ldw0,
        const __bf16* __restrict__ A1, int lda1, const __bf16* __restrict__ W1, int ldw1,
        const __bf16* __restrict__ A2, int lda2, const __bf16* __restrict__ W2, int ldw2,
        const float* __restrict__ bias,
        float* __restrict__ Cf, __bf16* __restrict__ Cb,
        int N, int K) {
    const int lane = threadIdx.x & 31;
    const int wave = threadIdx.x >> 5;
    const int half = lane >> 4;
    const int l15  = lane & 15;
    const int mTile = (blockIdx.x * 4 + wave) * 32;
    const int nTile = blockIdx.y * 64;

    v8f acc[8] = {};

    gemm_accum(A0, lda0, W0, ldw0, K, mTile, nTile, l15, half, acc);
    if (A1) gemm_accum(A1, lda1, W1, ldw1, K, mTile, nTile, l15, half, acc);
    if (A2) gemm_accum(A2, lda2, W2, ldw2, K, mTile, nTile, l15, half, acc);

    // C/D layout: VGPR r holds M = r + 8*half (+16*mi), N = l15 (+ 16*j + nTile)
    #pragma unroll
    for (int j = 0; j < 4; ++j) {
        int ncol = nTile + j * 16 + l15;
        float bv = bias ? bias[ncol] : 0.f;
        #pragma unroll
        for (int mi = 0; mi < 2; ++mi) {
            v8f a = acc[mi * 4 + j];
            #pragma unroll
            for (int r = 0; r < 8; ++r) {
                int m = mTile + mi * 16 + r + 8 * half;
                float val = a[r] + bv;
                if (Cf) Cf[(size_t)m * N + ncol] = val;
                if (Cb) Cb[(size_t)m * N + ncol] = (__bf16)val;
            }
        }
    }
}

// ---------------------------------------------------------------------------
// Kernel 3: fused slot scatter + softmax-weighted fusion.  Single sweep over
// slots (256MB in, 256MB out) -> the only HBM-bound pass.  Emits fused
// directly as bf16 (its only consumer is the next WMMA GEMM).
// ---------------------------------------------------------------------------
__global__ __launch_bounds__(256) void fuse_scatter_kernel(
        const float* __restrict__ slots,
        const float* __restrict__ v,
        const int*   __restrict__ ptr,
        const int*   __restrict__ mask,
        const float* __restrict__ w,
        const float* __restrict__ P,
        float* __restrict__ slots_out,
        __bf16* __restrict__ fused_bf) {
    const int b = blockIdx.y;
    const int d = blockIdx.x * 256 + threadIdx.x;
    const int p = ptr[b];
    const int m = mask[b];
    const size_t base = (size_t)b * S_SZ * D_SZ + d;
    const float vv = v[(size_t)b * D_SZ + d];
    float acc = P[d];
    #pragma unroll 8
    for (int s = 0; s < S_SZ; ++s) {
        float val = slots[base + (size_t)s * D_SZ];
        float nv  = (m && (s == p)) ? vv : val;
        slots_out[base + (size_t)s * D_SZ] = nv;
        acc += w[s] * nv;
    }
    fused_bf[(size_t)b * D_SZ + d] = (__bf16)acc;
}

// ---------------------------------------------------------------------------
// Kernel 4: elementwise LSTM tail.
// ---------------------------------------------------------------------------
__global__ __launch_bounds__(256) void lstm_kernel(
        const float* __restrict__ gates,
        const float* __restrict__ c_old,
        float* __restrict__ h_out,
        float* __restrict__ c_out) {
    const size_t idx = (size_t)blockIdx.x * 256 + threadIdx.x;   // B*H
    const size_t b = idx / H_SZ;
    const int    j = (int)(idx % H_SZ);
    const float* g = gates + b * 4 * H_SZ;
    float ig = g[j], fg = g[H_SZ + j], gg = g[2 * H_SZ + j], og = g[3 * H_SZ + j];
    float si = 1.f / (1.f + __expf(-ig));
    float sf = 1.f / (1.f + __expf(-fg));
    float so = 1.f / (1.f + __expf(-og));
    float tg = tanhf(gg);
    float c  = sf * c_old[idx] + si * tg;
    float h  = so * tanhf(c);
    c_out[idx] = c;
    h_out[idx] = h;
}

// ---------------------------------------------------------------------------
extern "C" void kernel_launch(void* const* d_in, const int* in_sizes, int n_in,
                              void* d_out, int out_size, void* d_ws, size_t ws_size,
                              hipStream_t stream) {
    (void)in_sizes; (void)n_in; (void)out_size; (void)ws_size;

    const float* x       = (const float*)d_in[0];   // (B,D)
    const float* h_lstm  = (const float*)d_in[1];   // (B,H)
    const float* c_lstm  = (const float*)d_in[2];   // (B,H)
    const float* slots   = (const float*)d_in[3];   // (B,S,D)
    const int*   ptr     = (const int*)  d_in[4];   // (B,)
    const float* Wv      = (const float*)d_in[5];   // (D,D)
    const float* bv      = (const float*)d_in[6];   // (D,)
    const float* det_w   = (const float*)d_in[7];   // (1,D)
    const float* det_b   = (const float*)d_in[8];   // (1,)
    const float* pos     = (const float*)d_in[9];   // (S,D)
    const float* mweights= (const float*)d_in[10];  // (S,)
    const float* Wproj   = (const float*)d_in[11];  // (H,D)
    const float* bproj   = (const float*)d_in[12];  // (H,)
    const float* Wih     = (const float*)d_in[13];  // (4H, D+H)
    const float* bih     = (const float*)d_in[14];  // (4H,)
    const float* Whh     = (const float*)d_in[15];  // (4H, H)

    const size_t BD = (size_t)B_SZ * D_SZ;      // 2,097,152
    const size_t BH = (size_t)B_SZ * H_SZ;      // 2,097,152
    const size_t BG = (size_t)B_SZ * 4 * H_SZ;  // 8,388,608
    const size_t WVN  = (size_t)D_SZ * D_SZ;            // 262,144
    const size_t WPN  = (size_t)H_SZ * D_SZ;            // 262,144
    const size_t WIHN = (size_t)(4 * H_SZ) * (D_SZ + H_SZ); // 2,097,152
    const size_t WHHN = (size_t)(4 * H_SZ) * H_SZ;      // 1,048,576

    // d_out = [ h_new (BH) | c_new (BH) | slots_new (B*S*D) | ptr_new (B) ]
    float* out_h     = (float*)d_out;
    float* out_c     = out_h + BH;
    float* out_slots = out_c + BH;
    int*   out_ptr   = (int*)(out_slots + (size_t)B_SZ * S_SZ * D_SZ);

    // Workspace carve-up (float region, then 16B-aligned bf16 region)
    float* ws_v     = (float*)d_ws;           // (B,D) float
    float* ws_gates = ws_v + BD;              // (B,4H) float
    float* ws_w     = ws_gates + BG;          // (S,)
    float* ws_P     = ws_w + 64;              // (D,)
    int*   ws_mask  = (int*)(ws_P + D_SZ);    // (B,)
    __bf16* bf_base  = (__bf16*)(ws_mask + B_SZ);
    __bf16* x_bf     = bf_base;               // (B,D)
    __bf16* hl_bf    = x_bf + BD;             // (B,H)
    __bf16* fused_bf = hl_bf + BH;            // (B,D)
    __bf16* hmem_bf  = fused_bf + BD;         // (B,H)
    __bf16* Wv_bf    = hmem_bf + BH;          // (D,D)
    __bf16* Wp_bf    = Wv_bf + WVN;           // (H,D)
    __bf16* Wih_bf   = Wp_bf + WPN;           // (4H, D+H)
    __bf16* Whh_bf   = Wih_bf + WIHN;         // (4H, H)

    // 0) one-time (per launch) f32 -> bf16 operand conversion (all L2-resident)
    auto cvt = [&](const float* in, __bf16* out, size_t n) {
        int n4 = (int)(n / 4);
        cvt_bf16_kernel<<<(n4 + 255) / 256, 256, 0, stream>>>(in, out, n4);
    };
    cvt(x, x_bf, BD);
    cvt(h_lstm, hl_bf, BH);
    cvt(Wv, Wv_bf, WVN);
    cvt(Wproj, Wp_bf, WPN);
    cvt(Wih, Wih_bf, WIHN);
    cvt(Whh, Whh_bf, WHHN);

    // 1) softmax(mem_weights), P[d]
    prep_kernel<<<1, 512, 0, stream>>>(mweights, pos, ws_w, ws_P);

    // 2) detector -> mask, ptr_new
    detector_kernel<<<B_SZ / 4, 128, 0, stream>>>(x, det_w, det_b, ptr, ws_mask, out_ptr);

    // 3) v = x @ Wv^T + bv      (M=4096, N=512, K=512) -> float
    wmma_gemm_kernel<<<dim3(B_SZ / 128, D_SZ / 64), 128, 0, stream>>>(
        x_bf, D_SZ, Wv_bf, D_SZ,
        nullptr, 0, nullptr, 0,
        nullptr, 0, nullptr, 0,
        bv, ws_v, nullptr, D_SZ, D_SZ);

    // 4) fused slot scatter + weighted fusion (single HBM sweep over slots)
    fuse_scatter_kernel<<<dim3(D_SZ / 256, B_SZ), 256, 0, stream>>>(
        slots, ws_v, ptr, ws_mask, ws_w, ws_P, out_slots, fused_bf);

    // 5) h_mem = fused @ Wproj^T + bproj   (M=4096, N=512, K=512) -> bf16
    wmma_gemm_kernel<<<dim3(B_SZ / 128, H_SZ / 64), 128, 0, stream>>>(
        fused_bf, D_SZ, Wp_bf, D_SZ,
        nullptr, 0, nullptr, 0,
        nullptr, 0, nullptr, 0,
        bproj, nullptr, hmem_bf, H_SZ, D_SZ);

    // 6) gates = x @ Wih[:, :D]^T + h_mem @ Wih[:, D:]^T + h_lstm @ Whh^T + bih
    //    (M=4096, N=2048, three K=512 pairs) -> float
    wmma_gemm_kernel<<<dim3(B_SZ / 128, (4 * H_SZ) / 64), 128, 0, stream>>>(
        x_bf,    D_SZ, Wih_bf,        D_SZ + H_SZ,
        hmem_bf, H_SZ, Wih_bf + D_SZ, D_SZ + H_SZ,
        hl_bf,   H_SZ, Whh_bf,        H_SZ,
        bih, ws_gates, nullptr, 4 * H_SZ, H_SZ);

    // 7) LSTM elementwise tail
    lstm_kernel<<<(unsigned)(BH / 256), 256, 0, stream>>>(ws_gates, c_lstm, out_h, out_c);
}
